// SingleRoundGraphLSTM_pyg_32933809225902
// MI455X (gfx1250) — compile-verified, matrix operand order checked by reading, not run on hardware
//
#include <hip/hip_runtime.h>
#include <hip/hip_bf16.h>

typedef __attribute__((ext_vector_type(16))) _Float16 v16h;
typedef __attribute__((ext_vector_type(8)))  _Float16 v8h;
typedef __attribute__((ext_vector_type(8)))  float    v8f;

#define NN   48000
#define RR   16
#define NPRn 3000
#define EPRn 6000
#define HHn  128

__device__ __forceinline__ float sigf(float x) { return 1.0f / (1.0f + expf(-x)); }

// Load one 16x32 f16 fragment (A layout; B^T stored row-major [out,K] loads identically)
// Lane L (row = L&15, kh = L>>4) holds K elems {kh*8 .. kh*8+7} in v16h[0..7]
// and {16+kh*8 .. 16+kh*8+7} in v16h[8..15]  -> two 16B loads.
__device__ __forceinline__ v16h load_frag(const _Float16* __restrict__ base, int ld,
                                          int row0, int k0, int lane) {
    const int r  = row0 + (lane & 15);
    const int kh = (lane >> 4) << 3;
    const _Float16* p = base + (long)r * ld + k0 + kh;
    v8h lo = *(const v8h*)(p);
    v8h hi = *(const v8h*)(p + 16);
    v16h out;
#pragma unroll
    for (int i = 0; i < 8; ++i) { out[i] = lo[i]; out[i + 8] = hi[i]; }
    return out;
}

// C[M,512] = A[M,K] (f16, row-major) x Bt[512,K]^T (f16, row-major [out,K]), f32 accum.
// Block: 256 threads = 8 waves; wave w computes rows [blockIdx.x*16, +16) x cols [w*64, +64).
template <int K>
__global__ __launch_bounds__(256) void gemm_f16_wmma(const _Float16* __restrict__ A,
                                                     const _Float16* __restrict__ Bt,
                                                     float* __restrict__ C) {
    const int lane = threadIdx.x & 31;
    const int wave = threadIdx.x >> 5;
    const int row0 = blockIdx.x * 16;
    const int col0 = wave * 64;
    v8f acc0 = {}, acc1 = {}, acc2 = {}, acc3 = {};
#pragma unroll
    for (int k0 = 0; k0 < K; k0 += 32) {
        v16h a  = load_frag(A, K, row0, k0, lane);
        v16h b0 = load_frag(Bt, K, col0 +  0, k0, lane);
        v16h b1 = load_frag(Bt, K, col0 + 16, k0, lane);
        v16h b2 = load_frag(Bt, K, col0 + 32, k0, lane);
        v16h b3 = load_frag(Bt, K, col0 + 48, k0, lane);
        acc0 = __builtin_amdgcn_wmma_f32_16x16x32_f16(false, a, false, b0, (short)0, acc0, false, false);
        acc1 = __builtin_amdgcn_wmma_f32_16x16x32_f16(false, a, false, b1, (short)0, acc1, false, false);
        acc2 = __builtin_amdgcn_wmma_f32_16x16x32_f16(false, a, false, b2, (short)0, acc2, false, false);
        acc3 = __builtin_amdgcn_wmma_f32_16x16x32_f16(false, a, false, b3, (short)0, acc3, false, false);
    }
    // D layout: v8f element v -> row = row0 + 8*(lane>>4) + v, col = col0 + (lane&15)
    const int ccol  = lane & 15;
    const int rbase = row0 + ((lane >> 4) << 3);
#pragma unroll
    for (int v = 0; v < 8; ++v) {
        long roff = (long)(rbase + v) * 512;
        C[roff + col0 +  0 + ccol] = acc0[v];
        C[roff + col0 + 16 + ccol] = acc1[v];
        C[roff + col0 + 32 + ccol] = acc2[v];
        C[roff + col0 + 48 + ccol] = acc3[v];
    }
}

__global__ void f32_to_f16_kernel(const float* __restrict__ src, _Float16* __restrict__ dst, int n) {
    int i = blockIdx.x * blockDim.x + threadIdx.x;
    if (i < n) dst[i] = (_Float16)src[i];
}

// Round 0: nodes [0, NPR): c = sig(i)*tanh(u); h = sig(o)*tanh(c)
__global__ __launch_bounds__(256) void round0_kernel(const float* __restrict__ P,
                                                     const float* __restrict__ b_iou,
                                                     float* __restrict__ cbuf,
                                                     float* __restrict__ hout,
                                                     _Float16* __restrict__ h16) {
    int tid  = blockIdx.x * blockDim.x + threadIdx.x;
    int node = tid >> 7, col = tid & 127;
    const float* pr = P + (long)node * 512;
    float iv = pr[col]        + b_iou[col];
    float ov = pr[128 + col]  + b_iou[128 + col];
    float uv = pr[256 + col]  + b_iou[256 + col];
    float cv = sigf(iv) * tanhf(uv);
    float hv = sigf(ov) * tanhf(cv);
    long o = (long)node * HHn + col;
    cbuf[o] = cv; hout[o] = hv; h16[o] = (_Float16)hv;
}

// Build per-round edge A matrix [EPR, 256] f16: row e = [(1-lab)*h_child | lab*h_child]
__global__ __launch_bounds__(256) void build_edge_A(const _Float16* __restrict__ h16,
                                                    const int* __restrict__ child,
                                                    const float* __restrict__ lab,
                                                    _Float16* __restrict__ Ae) {
    int tid = blockIdx.x * blockDim.x + threadIdx.x;
    int e = tid >> 7, col = tid & 127;
    float l  = lab[e];
    float hv = (float)h16[(long)child[e] * HHn + col];
    Ae[(long)e * 256 + col]       = (_Float16)((1.0f - l) * hv);
    Ae[(long)e * 256 + 128 + col] = (_Float16)(l * hv);
}

// Per-parent gate math for round r (parents r*NPR + j, edges 2j, 2j+1).
__global__ __launch_bounds__(256) void round_update(const float* __restrict__ P,
                                                    const float* __restrict__ Y,
                                                    const int* __restrict__ child,
                                                    const float* __restrict__ b_iou,
                                                    const float* __restrict__ b_f,
                                                    float* __restrict__ cbuf,
                                                    float* __restrict__ hout,
                                                    _Float16* __restrict__ h16,
                                                    int r) {
    int tid = blockIdx.x * blockDim.x + threadIdx.x;
    int j = tid >> 7, col = tid & 127;
    int p  = r * NPRn + j;
    int e0 = 2 * j, e1 = 2 * j + 1;
    int c0 = child[e0], c1 = child[e1];
    const float* pp = P + (long)p * 512;
    const float* y0 = Y + (long)e0 * 512;
    const float* y1 = Y + (long)e1 * 512;
    float iv = pp[col]        + y0[col]        + y1[col]        + b_iou[col];
    float ov = pp[128 + col]  + y0[128 + col]  + y1[128 + col]  + b_iou[128 + col];
    float uv = pp[256 + col]  + y0[256 + col]  + y1[256 + col]  + b_iou[256 + col];
    float f0 = sigf(P[(long)c0 * 512 + 384 + col] + y0[384 + col] + b_f[col]);
    float f1 = sigf(P[(long)c1 * 512 + 384 + col] + y1[384 + col] + b_f[col]);
    float cc0 = fminf(fmaxf(cbuf[(long)c0 * HHn + col], -1e14f), 1e14f);
    float cc1 = fminf(fmaxf(cbuf[(long)c1 * HHn + col], -1e14f), 1e14f);
    float cv = sigf(iv) * tanhf(uv) + f0 * cc0 + f1 * cc1;
    float hv = sigf(ov) * tanhf(cv);
    long o = (long)p * HHn + col;
    cbuf[o] = cv; hout[o] = hv; h16[o] = (_Float16)hv;
}

extern "C" void kernel_launch(void* const* d_in, const int* in_sizes, int n_in,
                              void* d_out, int out_size, void* d_ws, size_t ws_size,
                              hipStream_t stream) {
    const float* x      = (const float*)d_in[0];   // [N,128]
    const float* labels = (const float*)d_in[1];   // [15, EPR]
    const float* W_iou  = (const float*)d_in[2];   // [384,128]
    const float* W_f    = (const float*)d_in[3];   // [128,128]
    const float* b_iou  = (const float*)d_in[4];   // [384]
    const float* b_f    = (const float*)d_in[5];   // [128]
    const float* U_iou  = (const float*)d_in[6];   // [384,256]
    const float* U_f    = (const float*)d_in[7];   // [128,256]
    const int*   edges  = (const int*)d_in[8];     // [15, 2, EPR]
    float* hout = (float*)d_out;                   // [N,128] == final h (scatter is identity)

    // ---- workspace carve-up (256B aligned) ----
    size_t off = 0;
    auto alloc = [&](size_t bytes) -> void* {
        void* p = (char*)d_ws + off;
        off += (bytes + 255) & ~(size_t)255;
        return p;
    };
    _Float16* x16    = (_Float16*)alloc((size_t)NN * 128 * 2);
    _Float16* Wall16 = (_Float16*)alloc((size_t)512 * 128 * 2);   // [W_iou; W_f]
    _Float16* Uall16 = (_Float16*)alloc((size_t)512 * 256 * 2);   // [U_iou; U_f]
    float*    P      = (float*)   alloc((size_t)NN * 512 * 4);    // x @ [W_iou;W_f]^T
    float*    cbuf   = (float*)   alloc((size_t)NN * HHn * 4);
    _Float16* h16    = (_Float16*)alloc((size_t)NN * HHn * 2);
    _Float16* Ae     = (_Float16*)alloc((size_t)EPRn * 256 * 2);
    float*    Y      = (float*)   alloc((size_t)EPRn * 512 * 4);
    (void)ws_size; (void)in_sizes; (void)n_in; (void)out_size;

    // ---- f32 -> f16 conversions (x, packed weight matrices) ----
    {
        int n = NN * 128;
        f32_to_f16_kernel<<<(n + 255) / 256, 256, 0, stream>>>(x, x16, n);
        f32_to_f16_kernel<<<(384 * 128 + 255) / 256, 256, 0, stream>>>(W_iou, Wall16, 384 * 128);
        f32_to_f16_kernel<<<(128 * 128 + 255) / 256, 256, 0, stream>>>(W_f, Wall16 + 384 * 128, 128 * 128);
        f32_to_f16_kernel<<<(384 * 256 + 255) / 256, 256, 0, stream>>>(U_iou, Uall16, 384 * 256);
        f32_to_f16_kernel<<<(128 * 256 + 255) / 256, 256, 0, stream>>>(U_f, Uall16 + 384 * 256, 128 * 256);
    }

    // ---- P[N,512] = x @ [W_iou; W_f]^T  (WMMA, K=128) ----
    gemm_f16_wmma<128><<<NN / 16, 256, 0, stream>>>(x16, Wall16, P);

    // ---- round 0 (leaves) ----
    round0_kernel<<<(NPRn * HHn) / 256, 256, 0, stream>>>(P, b_iou, cbuf, hout, h16);

    // ---- rounds 1..15 (serial dependency) ----
    for (int r = 1; r < RR; ++r) {
        const int* child = edges + (size_t)(r - 1) * 2 * EPRn + EPRn;  // edge_index[1]
        const float* lab = labels + (size_t)(r - 1) * EPRn;
        build_edge_A<<<(EPRn * HHn) / 256, 256, 0, stream>>>(h16, child, lab, Ae);
        gemm_f16_wmma<256><<<EPRn / 16, 256, 0, stream>>>(Ae, Uall16, Y);
        round_update<<<(NPRn * HHn) / 256, 256, 0, stream>>>(P, Y, child, b_iou, b_f,
                                                             cbuf, hout, h16, r);
    }
}
